// GraphConv_5437428597451
// MI455X (gfx1250) — matrix operand order repaired and checked by reading, not compiled, not run on hardware
//
#include <hip/hip_runtime.h>
#include <cstddef>

#define NN 8192
#define SPLITK 4

typedef __attribute__((ext_vector_type(16))) __bf16          v16bf;
typedef __attribute__((ext_vector_type(8)))  float           v8f;
typedef __attribute__((ext_vector_type(4)))  float           v4f;
typedef __attribute__((ext_vector_type(8)))  unsigned short  v8us;

// f32 -> bf16: native fptrunc (RNE); lets the backend use packed cvt instructions
__device__ __forceinline__ __bf16 f2bf(float f) { return (__bf16)f; }

union FragU {
  v16bf  v;
  __bf16 s[16];
  v8us   h[2];
};

// ---------------------------------------------------------------------------
// Pack X [NN x K] f32 row-major into WMMA B-fragment layout (bf16).
// Fragment (kt, nt): 32 lanes x 16 bf16; frag[lane][e] = H[kt*32 + (lane>>4)*16 + e][nt*16 + (lane&15)]
// ---------------------------------------------------------------------------
template <int K>
__global__ __launch_bounds__(256) void pack_kernel(const float* __restrict__ X,
                                                   unsigned short* __restrict__ P) {
  constexpr int NT = K / 16;
  int t    = blockIdx.x * 256 + threadIdx.x;      // one task = one lane of one fragment
  int lane = t & 31;
  int f    = t >> 5;
  if (f >= (NN / 32) * NT) return;
  int kt   = f / NT;
  int nt   = f % NT;
  int half = lane >> 4;
  int col  = nt * 16 + (lane & 15);
  FragU pk;
#pragma unroll
  for (int e = 0; e < 16; ++e) {
    int r = kt * 32 + half * 16 + e;
    pk.s[e] = f2bf(X[(size_t)r * K + col]);
  }
  unsigned short* dst = P + (size_t)f * 512 + lane * 16;
  *(v8us*)(dst)     = pk.h[0];
  *(v8us*)(dst + 8) = pk.h[1];
}

// ---------------------------------------------------------------------------
// Split-K GEMM: Cpart[s] = A[:, kc0:kc0+NN/SPLITK] @ H[kc0:kc0+NN/SPLITK, :]
// 8 waves per block, each wave owns a 16-row strip; bf16 WMMA 16x16x32, f32 acc.
// Layer 1 (A_F32) converts A to bf16 in-register and optionally caches it.
// ---------------------------------------------------------------------------
template <int K_IN, bool A_F32, bool CACHE_A>
__global__ __launch_bounds__(256) void gcn_gemm(const float* __restrict__ A32,
                                                const unsigned short* __restrict__ A16,
                                                const unsigned short* __restrict__ Hpack,
                                                float* __restrict__ Cpart,
                                                unsigned short* __restrict__ AbfOut) {
  constexpr int NT = K_IN / 16;
  const int lane  = threadIdx.x & 31;
  const int wave  = threadIdx.x >> 5;
  const int strip = blockIdx.x * 8 + wave;     // 0 .. NN/16-1
  const int r0    = strip * 16;
  const int s     = blockIdx.y;                // split-K index
  const int kc0   = s * (NN / SPLITK);
  const int half  = lane >> 4;

  v8f acc[NT] = {};

  const size_t arow = (size_t)(r0 + (lane & 15)) * NN;   // A row for this lane (A-fragment)

  for (int k0 = kc0; k0 < kc0 + NN / SPLITK; k0 += 32) {
    // --- B fragments: pre-packed, one 32B contiguous load per lane per tile ---
    const unsigned short* hp = Hpack + ((size_t)(k0 >> 5) * NT) * 512 + lane * 16;
    v16bf b[NT];
#pragma unroll
    for (int nt = 0; nt < NT; ++nt)
      b[nt] = *(const v16bf*)(hp + nt * 512);

    // --- A fragment: lane(half,row) holds K = {half*16+0..15} ---
    FragU au;
    if constexpr (A_F32) {
      const float* ap = A32 + arow + k0 + half * 16;
      __builtin_prefetch(ap + 512, 0, 0);                // stream A ahead (global_prefetch)
      v4f l0 = *(const v4f*)(ap + 0);
      v4f l1 = *(const v4f*)(ap + 4);
      v4f h0 = *(const v4f*)(ap + 8);
      v4f h1 = *(const v4f*)(ap + 12);
#pragma unroll
      for (int e = 0; e < 4; ++e) {
        au.s[e]      = f2bf(l0[e]);
        au.s[4 + e]  = f2bf(l1[e]);
        au.s[8 + e]  = f2bf(h0[e]);
        au.s[12 + e] = f2bf(h1[e]);
      }
      if constexpr (CACHE_A) {                            // persist bf16 A for layers 2-4
        unsigned short* dst = AbfOut + arow + k0 + half * 16;
        *(v8us*)(dst)      = au.h[0];
        *(v8us*)(dst + 8)  = au.h[1];
      }
    } else {
      const unsigned short* ap = A16 + arow + k0 + half * 16;
      __builtin_prefetch(ap + 1024, 0, 0);
      au.h[0] = *(const v8us*)(ap);
      au.h[1] = *(const v8us*)(ap + 8);
    }

    // --- matrix core ---
#pragma unroll
    for (int nt = 0; nt < NT; ++nt)
      acc[nt] = __builtin_amdgcn_wmma_f32_16x16x32_bf16(
          false, au.v, false, b[nt], (short)0, acc[nt], false, false);
  }

  // D layout: VGPR v, lane(half,n) -> C[r0 + v + 8*half][nt*16 + n]
  const size_t sbase = (size_t)s * NN;
#pragma unroll
  for (int nt = 0; nt < NT; ++nt) {
#pragma unroll
    for (int v = 0; v < 8; ++v) {
      int r = r0 + v + 8 * half;
      Cpart[(sbase + r) * K_IN + nt * 16 + (lane & 15)] = acc[nt][v];
    }
  }
}

// ---------------------------------------------------------------------------
// Deterministic split-K reduction + fused  (.)@W^T  (+ReLU)  epilogue.
// Each block handles 64 rows; writes next layer's packed-bf16 H, or f32 out.
// ---------------------------------------------------------------------------
template <int K_IN, int K_OUT, bool RELU, bool LAST>
__global__ __launch_bounds__(256) void reduce_epilogue(const float* __restrict__ Cpart,
                                                       const float* __restrict__ W,
                                                       unsigned short* __restrict__ HpackOut,
                                                       float* __restrict__ Out) {
  __shared__ float Cs[64][K_IN];
  __shared__ float Ws[K_OUT][K_IN];
  const int tid = threadIdx.x;
  const int rb0 = blockIdx.x * 64;

  for (int i = tid; i < K_OUT * K_IN; i += 256)
    Ws[i / K_IN][i % K_IN] = W[i];

  for (int i = tid; i < 64 * K_IN; i += 256) {
    int rl = i / K_IN, c = i % K_IN;
    float sum = 0.f;
#pragma unroll
    for (int s2 = 0; s2 < SPLITK; ++s2)                 // fixed order -> deterministic
      sum += Cpart[((size_t)s2 * NN + rb0 + rl) * K_IN + c];
    Cs[rl][c] = sum;
  }
  __syncthreads();

  // one task = 16 consecutive rows x 1 output column (matches one pack-lane chunk)
  constexpr int NTASKS = 4 * K_OUT;
  if (tid < NTASKS) {
    int qt  = tid / K_OUT;
    int o   = tid % K_OUT;
    int lrb = qt * 16;
    float v[16];
#pragma unroll
    for (int e = 0; e < 16; ++e) {
      float a2 = 0.f;
#pragma unroll
      for (int c = 0; c < K_IN; ++c)
        a2 += Cs[lrb + e][c] * Ws[o][c];
      v[e] = RELU ? fmaxf(a2, 0.f) : a2;
    }
    if constexpr (!LAST) {
      int R     = rb0 + lrb;                            // multiple of 16
      int kt    = R >> 5;
      int half2 = (R >> 4) & 1;
      int nt    = o >> 4;
      int lane2 = half2 * 16 + (o & 15);
      FragU pk;
#pragma unroll
      for (int e = 0; e < 16; ++e) pk.s[e] = f2bf(v[e]);
      unsigned short* dst =
          HpackOut + ((size_t)kt * (K_OUT / 16) + nt) * 512 + lane2 * 16;
      *(v8us*)(dst)     = pk.h[0];
      *(v8us*)(dst + 8) = pk.h[1];
    } else {
#pragma unroll
      for (int e = 0; e < 16; ++e)
        Out[(size_t)(rb0 + lrb + e) * K_OUT + o] = v[e];
    }
  }
}

// ---------------------------------------------------------------------------
extern "C" void kernel_launch(void* const* d_in, const int* in_sizes, int n_in,
                              void* d_out, int out_size, void* d_ws, size_t ws_size,
                              hipStream_t stream) {
  (void)in_sizes; (void)n_in; (void)out_size;
  const float* A  = (const float*)d_in[0];
  const float* X  = (const float*)d_in[1];
  const float* W1 = (const float*)d_in[2];
  const float* W2 = (const float*)d_in[3];
  const float* W3 = (const float*)d_in[4];
  const float* W4 = (const float*)d_in[5];
  float* out = (float*)d_out;

  char* ws = (char*)d_ws;
  unsigned short* pack0 = (unsigned short*)(ws);                 // 1 MB
  unsigned short* pack1 = (unsigned short*)(ws + (1u << 20));    // 1 MB
  float* Cpart = (float*)(ws + (2u << 20));                      // SPLITK * NN * 64 * 4 = 8 MB
  const size_t baseNeed = (size_t)(2u << 20) + (size_t)SPLITK * NN * 64 * sizeof(float);
  unsigned short* Abf = nullptr;
  const bool cacheA = (ws_size >= baseNeed + (size_t)NN * NN * sizeof(unsigned short));
  if (cacheA) Abf = (unsigned short*)(ws + baseNeed);

  dim3 gg(NN / 128, SPLITK);

  // pack X into B-fragment layout
  pack_kernel<32><<<((NN / 32) * 2 * 32) / 256, 256, 0, stream>>>(X, pack0);

  // Layer 1: h1 = relu((A@X) @ W1^T), also cache bf16 A if workspace allows
  if (cacheA)
    gcn_gemm<32, true, true><<<gg, 256, 0, stream>>>(A, nullptr, pack0, Cpart, Abf);
  else
    gcn_gemm<32, true, false><<<gg, 256, 0, stream>>>(A, nullptr, pack0, Cpart, nullptr);
  reduce_epilogue<32, 32, true, false><<<NN / 64, 256, 0, stream>>>(Cpart, W1, pack1, nullptr);

  // Layer 2: h2 = relu((A@h1) @ W2^T)
  if (cacheA)
    gcn_gemm<32, false, false><<<gg, 256, 0, stream>>>(nullptr, Abf, pack1, Cpart, nullptr);
  else
    gcn_gemm<32, true, false><<<gg, 256, 0, stream>>>(A, nullptr, pack1, Cpart, nullptr);
  reduce_epilogue<32, 64, true, false><<<NN / 64, 256, 0, stream>>>(Cpart, W2, pack0, nullptr);

  // Layer 3: h3 = relu((A@h2) @ W3^T)
  if (cacheA)
    gcn_gemm<64, false, false><<<gg, 256, 0, stream>>>(nullptr, Abf, pack0, Cpart, nullptr);
  else
    gcn_gemm<64, true, false><<<gg, 256, 0, stream>>>(A, nullptr, pack0, Cpart, nullptr);
  reduce_epilogue<64, 64, true, false><<<NN / 64, 256, 0, stream>>>(Cpart, W3, pack1, nullptr);

  // Layer 4: out = (A@h3) @ W4^T  (no relu)
  if (cacheA)
    gcn_gemm<64, false, false><<<gg, 256, 0, stream>>>(nullptr, Abf, pack1, Cpart, nullptr);
  else
    gcn_gemm<64, true, false><<<gg, 256, 0, stream>>>(A, nullptr, pack1, Cpart, nullptr);
  reduce_epilogue<64, 32, false, true><<<NN / 64, 256, 0, stream>>>(Cpart, W4, nullptr, out);
}
